// Net_11192684773571
// MI455X (gfx1250) — matrix operand order repaired and checked by reading, not compiled, not run on hardware
//
#include <hip/hip_runtime.h>

// ---------------- model constants ----------------
#define B_    2
#define L_    2048
#define NROW  (B_ * L_)        // 4096 token rows
#define INDIM 16
#define DM    512              // d_model
#define DI    1024             // d_inner
#define DSTATE 16
#define DTRANK 32
#define DBCW  (DTRANK + 2 * DSTATE)   // 64
#define NLAYERS 2

// GEMM tiling
#define BM 128
#define BN 64
#define KSTEP 64               // staged K per TDM transfer (2 WMMA k-chunks)

typedef __bf16 bf16_t;
typedef __attribute__((ext_vector_type(16))) __bf16 v16bf;
typedef __attribute__((ext_vector_type(8)))  float  v8f;
typedef __attribute__((ext_vector_type(4)))  unsigned int v4u;
typedef __attribute__((ext_vector_type(8)))  int  v8i_t;
typedef __attribute__((ext_vector_type(4)))  int  v4i_t;

union FragU {
    v16bf v;
    unsigned int  u[8];
    uint4         u4[2];
    unsigned short s[16];
};

// fp32 -> bf16 round-to-nearest-even (bit pattern)
__device__ __forceinline__ unsigned short f2bf(float x) {
    unsigned int u = __float_as_uint(x);
    u += 0x7FFFu + ((u >> 16) & 1u);
    return (unsigned short)(u >> 16);
}

__device__ __forceinline__ float silu(float x) {
    return x / (1.f + __expf(-x));
}

// ---------------- fp32 -> bf16 convert ----------------
__global__ void cvt_bf16_kernel(const float* __restrict__ in,
                                unsigned short* __restrict__ out, int n) {
    int i = blockIdx.x * blockDim.x + threadIdx.x;
    if (i < n) out[i] = f2bf(in[i]);
}

// ---------------- lin1: h = x @ W1^T + b1  (K=16, VALU) ----------------
__global__ void lin1_kernel(const float* __restrict__ x,
                            const float* __restrict__ w,
                            const float* __restrict__ b,
                            float* __restrict__ h) {
    int idx = blockIdx.x * blockDim.x + threadIdx.x;   // NROW*DM
    if (idx >= NROW * DM) return;
    int n = idx & (DM - 1);
    int r = idx >> 9;
    const float* xr = x + r * INDIM;
    const float* wr = w + n * INDIM;
    float acc = b[n];
#pragma unroll
    for (int k = 0; k < INDIM; ++k) acc += xr[k] * wr[k];
    h[idx] = acc;
}

// ---------------- rmsnorm + bf16 cast (one block per row) ----------------
__global__ void rmsnorm_bf16_kernel(const float* __restrict__ h,
                                    const float* __restrict__ w,
                                    unsigned short* __restrict__ out) {
    int row = blockIdx.x;
    const float* p = h + row * DM;
    float ss = 0.f;
    for (int i = threadIdx.x; i < DM; i += 256) { float v = p[i]; ss += v * v; }
#pragma unroll
    for (int off = 16; off >= 1; off >>= 1) ss += __shfl_xor(ss, off, 32);
    __shared__ float sred[8];
    if ((threadIdx.x & 31) == 0) sred[threadIdx.x >> 5] = ss;
    __syncthreads();
    float tot = 0.f;
#pragma unroll
    for (int i = 0; i < 8; ++i) tot += sred[i];
    float inv = rsqrtf(tot / (float)DM + 1e-5f);
    for (int i = threadIdx.x; i < DM; i += 256)
        out[row * DM + i] = f2bf(p[i] * inv * w[i]);
}

// ---------------- WMMA fragment gathers (from LDS tile) ----------------
// A fragment: 16x32 bf16, lane m = lane&15; lanes>=16 get K offset +8/+24.
__device__ __forceinline__ v16bf load_a_frag(const unsigned short* __restrict__ A,
                                             int lda, int m0, int k0, int lane) {
    int m  = m0 + (lane & 15);
    int kb = k0 + ((lane & 16) ? 8 : 0);
    const unsigned short* p = A + m * lda;
    FragU f;
    f.u4[0] = *(const uint4*)(p + kb);
    f.u4[1] = *(const uint4*)(p + kb + 16);
    return f.v;
}

// B fragment: 32x16 bf16, lane n = lane&15; lanes 0-15 K=0..7,16..23, lanes 16-31 K=8..15,24..31.
__device__ __forceinline__ v16bf load_b_frag(const unsigned short* __restrict__ Bw,
                                             int ldb, int n0, int k0, int lane) {
    int n  = n0 + (lane & 15);
    int kk = k0 + ((lane & 16) ? 8 : 0);
    const unsigned short* p = Bw + n * ldb + kk;
    FragU f;
    f.u4[0] = *(const uint4*)(p);
    f.u4[1] = *(const uint4*)(p + 16);
    return f.v;
}

__device__ __forceinline__ void store_tile(float* __restrict__ C,
                                           const float* __restrict__ addC,
                                           int ldc, int m0, int n0, v8f acc, int lane) {
    int n  = n0 + (lane & 15);
    int mb = m0 + ((lane & 16) ? 8 : 0);
#pragma unroll
    for (int v = 0; v < 8; ++v) {
        int m = mb + v;
        float val = acc[v];
        if (addC) val += addC[m * ldc + n];
        C[m * ldc + n] = val;
    }
}

// ---------------- TDM: DMA a 2D bf16 tile (rows x KSTEP) into LDS ----------
// D# per CDNA5 ISA. data_size=2B. Reads past
// tensor_dim0 return zero (used to zero-pad the K=32 dt_proj GEMM).
__device__ __forceinline__ void tdm_load_2d(unsigned lds_byte_addr,
                                            const unsigned short* gptr,  // tile start
                                            unsigned rows,               // tile_dim1
                                            unsigned tdim0,              // valid K elems
                                            unsigned tdim1,              // tensor rows
                                            unsigned stride0)            // row stride (elems)
{
    unsigned long long ga = (unsigned long long)gptr;
    v4u g0;
    g0.x = 1u;                                            // count=1 (valid user D#)
    g0.y = lds_byte_addr;                                 // LDS dest (bytes)
    g0.z = (unsigned)(ga & 0xFFFFFFFFu);                  // global_addr[31:0]
    g0.w = (unsigned)((ga >> 32) & 0x01FFFFFFu) | (2u << 30);  // addr[56:32] | type=2
    v8i_t g1;
    g1[0] = (int)(1u << 16);                              // wg_mask=0, data_size=2B
    g1[1] = (int)((tdim0 & 0xFFFFu) << 16);               // tensor_dim0[15:0]
    g1[2] = (int)((tdim0 >> 16) | ((tdim1 & 0xFFFFu) << 16));  // td0 hi | td1 lo
    g1[3] = (int)((tdim1 >> 16) | ((unsigned)KSTEP << 16));    // td1 hi | tile_dim0
    g1[4] = (int)rows;                                    // tile_dim1 (tile_dim2=0)
    g1[5] = (int)stride0;                                 // tensor_dim0_stride[31:0]
    g1[6] = 0;
    g1[7] = 0;
    v4i_t z4 = {0, 0, 0, 0};
    v8i_t z8 = {0, 0, 0, 0, 0, 0, 0, 0};
    // amdgpu-toolchain (clang-23) 6-arg form: (g0, g1, g2, g3, extra, cpol)
    __builtin_amdgcn_tensor_load_to_lds(g0, g1, z4, z4, z8, 0);
}

// ---------------- WMMA bf16 NT GEMM with TDM double buffering ----------------
// C[M,N] = A[M,K] * B[N,K]^T (+addC). Block: 8 waves -> 128x64 tile,
// each wave 32x32. K staged in KSTEP chunks via tensor_load_to_lds.
__global__ void gemm_nt_bf16_tdm(const unsigned short* __restrict__ A,
                                 const unsigned short* __restrict__ Bw,
                                 const float* __restrict__ addC,
                                 float* __restrict__ C,
                                 int M, int N, int K) {
    __shared__ unsigned short sA[2][BM * KSTEP];   // 2 x 16 KB
    __shared__ unsigned short sB[2][BN * KSTEP];   // 2 x  8 KB
    int lane = threadIdx.x & 31;
    int w    = threadIdx.x >> 5;        // 0..7
    int wrow = w >> 1;                  // 0..3 -> +32*wrow in M
    int wcol = w & 1;                   // 0..1 -> +32*wcol in N
    int tm   = blockIdx.y * BM;
    int tn   = blockIdx.x * BN;

    // Flat addr of __shared__ has the LDS byte offset in its low 32 bits.
    unsigned ldsA[2] = { (unsigned)(unsigned long long)(void*)&sA[0][0],
                         (unsigned)(unsigned long long)(void*)&sA[1][0] };
    unsigned ldsB[2] = { (unsigned)(unsigned long long)(void*)&sB[0][0],
                         (unsigned)(unsigned long long)(void*)&sB[1][0] };

    int nk = (K + KSTEP - 1) / KSTEP;

    if (threadIdx.x < 32) {   // wave 0 drives the TDM pipeline
        tdm_load_2d(ldsA[0], A + (size_t)tm * K, BM, (unsigned)K, (unsigned)M, (unsigned)K);
        tdm_load_2d(ldsB[0], Bw + (size_t)tn * K, BN, (unsigned)K, (unsigned)N, (unsigned)K);
    }

    v8f a00 = {}, a01 = {}, a10 = {}, a11 = {};
    for (int ks = 0; ks < nk; ++ks) {
        int cur = ks & 1;
        if (threadIdx.x < 32) {
            if (ks + 1 < nk) {
                int k0n = (ks + 1) * KSTEP;
                unsigned kleft = (unsigned)(K - k0n);          // valid K in this stage
                unsigned kval  = kleft > KSTEP ? KSTEP : kleft;
                tdm_load_2d(ldsA[cur ^ 1], A + (size_t)tm * K + k0n, BM, kval, (unsigned)M, (unsigned)K);
                tdm_load_2d(ldsB[cur ^ 1], Bw + (size_t)tn * K + k0n, BN, kval, (unsigned)N, (unsigned)K);
                __builtin_amdgcn_s_wait_tensorcnt(2);   // current pair complete
            } else {
                __builtin_amdgcn_s_wait_tensorcnt(0);
            }
        }
        __syncthreads();                                // publish staged tiles
        const unsigned short* pa = &sA[cur][0];
        const unsigned short* pb = &sB[cur][0];
#pragma unroll
        for (int kc = 0; kc < KSTEP; kc += 32) {
            v16bf fa0 = load_a_frag(pa, KSTEP, wrow * 32,      kc, lane);
            v16bf fa1 = load_a_frag(pa, KSTEP, wrow * 32 + 16, kc, lane);
            v16bf fb0 = load_b_frag(pb, KSTEP, wcol * 32,      kc, lane);
            v16bf fb1 = load_b_frag(pb, KSTEP, wcol * 32 + 16, kc, lane);
            a00 = __builtin_amdgcn_wmma_f32_16x16x32_bf16(false, fa0, false, fb0, (short)0, a00, false, false);
            a01 = __builtin_amdgcn_wmma_f32_16x16x32_bf16(false, fa0, false, fb1, (short)0, a01, false, false);
            a10 = __builtin_amdgcn_wmma_f32_16x16x32_bf16(false, fa1, false, fb0, (short)0, a10, false, false);
            a11 = __builtin_amdgcn_wmma_f32_16x16x32_bf16(false, fa1, false, fb1, (short)0, a11, false, false);
        }
        __syncthreads();                                // done reading before overwrite
    }

    int m0 = tm + wrow * 32;
    int n0 = tn + wcol * 32;
    store_tile(C, addC, N, m0,      n0,      a00, lane);
    store_tile(C, addC, N, m0,      n0 + 16, a01, lane);
    store_tile(C, addC, N, m0 + 16, n0,      a10, lane);
    store_tile(C, addC, N, m0 + 16, n0 + 16, a11, lane);
}

// ---------------- causal depthwise conv (D_CONV=4) + SiLU ----------------
__global__ void conv_silu_kernel(const float* __restrict__ xz,      // [NROW, 2*DI]
                                 const float* __restrict__ cw,      // [DI, 4]
                                 const float* __restrict__ cb,      // [DI]
                                 float* __restrict__ u,             // [NROW, DI]
                                 unsigned short* __restrict__ ubf)  // [NROW, DI]
{
    int idx = blockIdx.x * blockDim.x + threadIdx.x;   // NROW*DI
    if (idx >= NROW * DI) return;
    int d = idx & (DI - 1);
    int r = idx >> 10;
    int l = r & (L_ - 1);
    const float* w4 = cw + d * 4;
    float acc = cb[d];
#pragma unroll
    for (int j = 0; j < 4; ++j) {
        int ls = l - 3 + j;
        if (ls >= 0) acc += w4[j] * xz[(r - 3 + j) * (2 * DI) + d];
    }
    float y = silu(acc);
    u[idx]   = y;
    ubf[idx] = f2bf(y);
}

// ---------------- split dbc[:, :DTRANK] -> bf16 ----------------
__global__ void split_dt_kernel(const float* __restrict__ dbc,
                                unsigned short* __restrict__ dR) {
    int idx = blockIdx.x * blockDim.x + threadIdx.x;   // NROW*DTRANK
    if (idx >= NROW * DTRANK) return;
    int r = idx >> 5, j = idx & 31;
    dR[idx] = f2bf(dbc[r * DBCW + j]);
}

// ---------------- softplus(delta + dtb) in place ----------------
__global__ void softplus_kernel(float* __restrict__ delta,
                                const float* __restrict__ dtb) {
    int idx = blockIdx.x * blockDim.x + threadIdx.x;   // NROW*DI
    if (idx >= NROW * DI) return;
    int d = idx & (DI - 1);
    float xv = delta[idx] + dtb[d];
    delta[idx] = (xv > 20.f) ? xv : log1pf(__expf(xv));
}

// ---------------- selective scan: 16 lanes per (b,d) channel ----------------
__global__ void scan_kernel(const float* __restrict__ delta,  // [NROW, DI]
                            const float* __restrict__ u,      // [NROW, DI]
                            const float* __restrict__ dbc,    // [NROW, DBCW]
                            const float* __restrict__ A_log,  // [DI, DSTATE]
                            const float* __restrict__ Dp,     // [DI]
                            float* __restrict__ ys)           // [NROW, DI]
{
    int tid = threadIdx.x;
    int n   = tid & 15;
    int ch  = blockIdx.x * 16 + (tid >> 4);   // blockDim = 256 -> 16 channels
    int b   = ch >> 10;
    int d   = ch & (DI - 1);
    float Aval = -__expf(A_log[d * DSTATE + n]);
    float Dval = Dp[d];
    float hst  = 0.f;
    int rbase  = b * L_;
    for (int t = 0; t < L_; ++t) {
        int r = rbase + t;
        float dl = delta[r * DI + d];
        float uu = u[r * DI + d];
        float bm = dbc[r * DBCW + DTRANK + n];
        float cc = dbc[r * DBCW + DTRANK + DSTATE + n];
        float dA = __expf(dl * Aval);
        hst = dA * hst + (dl * uu) * bm;
        float part = hst * cc;
#pragma unroll
        for (int off = 8; off >= 1; off >>= 1) part += __shfl_xor(part, off, 32);
        if (n == 0) ys[r * DI + d] = part + uu * Dval;
    }
}

// ---------------- gate: yg = bf16(ys * silu(z)) ----------------
__global__ void gate_kernel(const float* __restrict__ ys,
                            const float* __restrict__ xz,
                            unsigned short* __restrict__ yg) {
    int idx = blockIdx.x * blockDim.x + threadIdx.x;   // NROW*DI
    if (idx >= NROW * DI) return;
    int d = idx & (DI - 1);
    int r = idx >> 10;
    float z = xz[r * (2 * DI) + DI + d];
    yg[idx] = f2bf(ys[idx] * silu(z));
}

// ---------------- lin2 + sigmoid (wave per row) ----------------
__global__ void lin2_kernel(const float* __restrict__ h,
                            const float* __restrict__ w,
                            const float* __restrict__ b,
                            float* __restrict__ out) {
    int row  = blockIdx.x * (blockDim.x >> 5) + (threadIdx.x >> 5);
    if (row >= NROW) return;
    int lane = threadIdx.x & 31;
    const float* hr = h + row * DM;
    float acc = 0.f;
    for (int i = lane; i < DM; i += 32) acc += hr[i] * w[i];
#pragma unroll
    for (int off = 16; off >= 1; off >>= 1) acc += __shfl_xor(acc, off, 32);
    if (lane == 0) {
        float s = acc + b[0];
        out[row] = 1.f / (1.f + __expf(-s));
    }
}

// ---------------- host-side launch ----------------
static inline void launch_gemm(const unsigned short* A, const unsigned short* Bw,
                               const float* addC, float* C,
                               int M, int N, int K, hipStream_t s) {
    dim3 grid((unsigned)(N / BN), (unsigned)(M / BM));
    gemm_nt_bf16_tdm<<<grid, 256, 0, s>>>(A, Bw, addC, C, M, N, K);
}

extern "C" void kernel_launch(void* const* d_in, const int* in_sizes, int n_in,
                              void* d_out, int out_size, void* d_ws, size_t ws_size,
                              hipStream_t stream) {
    const float* x        = (const float*)d_in[0];
    const float* lin1_w   = (const float*)d_in[1];
    const float* lin1_b   = (const float*)d_in[2];
    const float* lin2_w   = (const float*)d_in[3];
    const float* lin2_b   = (const float*)d_in[4];
    const float* norm_w   = (const float*)d_in[5];
    const float* in_proj  = (const float*)d_in[6];
    const float* conv_w   = (const float*)d_in[7];
    const float* conv_b   = (const float*)d_in[8];
    const float* x_proj   = (const float*)d_in[9];
    const float* dt_proj  = (const float*)d_in[10];
    const float* dt_b     = (const float*)d_in[11];
    const float* A_log    = (const float*)d_in[12];
    const float* D_param  = (const float*)d_in[13];
    const float* out_proj = (const float*)d_in[14];
    float* out = (float*)d_out;

    char* ws = (char*)d_ws;
    size_t off = 0;
    auto alloc = [&](size_t bytes) -> void* {
        void* p = ws + off;
        off += (bytes + 255) & ~(size_t)255;
        return p;
    };

    float*          h      = (float*)alloc((size_t)NROW * DM * 4);
    unsigned short* hn     = (unsigned short*)alloc((size_t)NROW * DM * 2);
    float*          xz     = (float*)alloc((size_t)NROW * 2 * DI * 4);
    float*          u      = (float*)alloc((size_t)NROW * DI * 4);
    unsigned short* xinbf  = (unsigned short*)alloc((size_t)NROW * DI * 2);  // reused as yg
    float*          dbc    = (float*)alloc((size_t)NROW * DBCW * 4);
    unsigned short* dR     = (unsigned short*)alloc((size_t)NROW * DTRANK * 2);
    float*          delta  = (float*)alloc((size_t)NROW * DI * 4);
    float*          ys     = (float*)alloc((size_t)NROW * DI * 4);
    unsigned short* w_in   = (unsigned short*)alloc((size_t)NLAYERS * 2 * DI * DM * 2);
    unsigned short* w_xp   = (unsigned short*)alloc((size_t)NLAYERS * DBCW * DI * 2);
    unsigned short* w_dt   = (unsigned short*)alloc((size_t)NLAYERS * DI * DTRANK * 2);
    unsigned short* w_out  = (unsigned short*)alloc((size_t)NLAYERS * DM * DI * 2);

    // --- weight fp32 -> bf16 ---
    {
        int n;
        n = NLAYERS * 2 * DI * DM;
        cvt_bf16_kernel<<<(n + 255) / 256, 256, 0, stream>>>(in_proj, w_in, n);
        n = NLAYERS * DBCW * DI;
        cvt_bf16_kernel<<<(n + 255) / 256, 256, 0, stream>>>(x_proj, w_xp, n);
        n = NLAYERS * DI * DTRANK;
        cvt_bf16_kernel<<<(n + 255) / 256, 256, 0, stream>>>(dt_proj, w_dt, n);
        n = NLAYERS * DM * DI;
        cvt_bf16_kernel<<<(n + 255) / 256, 256, 0, stream>>>(out_proj, w_out, n);
    }

    // --- lin1 ---
    lin1_kernel<<<(NROW * DM) / 256, 256, 0, stream>>>(x, lin1_w, lin1_b, h);

    // --- mamba layers ---
    for (int i = 0; i < NLAYERS; ++i) {
        rmsnorm_bf16_kernel<<<NROW, 256, 0, stream>>>(h, norm_w + i * DM, hn);

        // xz = hn @ in_proj^T     [NROW, 2*DI]
        launch_gemm(hn, w_in + (size_t)i * 2 * DI * DM, nullptr, xz,
                    NROW, 2 * DI, DM, stream);

        conv_silu_kernel<<<(NROW * DI) / 256, 256, 0, stream>>>(
            xz, conv_w + (size_t)i * DI * 4, conv_b + (size_t)i * DI, u, xinbf);

        // dbc = xin @ x_proj^T    [NROW, 64]
        launch_gemm(xinbf, w_xp + (size_t)i * DBCW * DI, nullptr, dbc,
                    NROW, DBCW, DI, stream);

        split_dt_kernel<<<(NROW * DTRANK) / 256, 256, 0, stream>>>(dbc, dR);

        // delta = dR @ dt_proj^T  [NROW, DI]  (K=32 zero-padded to KSTEP by TDM OOB)
        launch_gemm(dR, w_dt + (size_t)i * DI * DTRANK, nullptr, delta,
                    NROW, DI, DTRANK, stream);

        softplus_kernel<<<(NROW * DI) / 256, 256, 0, stream>>>(delta, dt_b + (size_t)i * DI);

        scan_kernel<<<(B_ * DI) / 16, 256, 0, stream>>>(
            delta, u, dbc, A_log + (size_t)i * DI * DSTATE, D_param + (size_t)i * DI, ys);

        gate_kernel<<<(NROW * DI) / 256, 256, 0, stream>>>(ys, xz, xinbf);

        // h = h + yg @ out_proj^T
        launch_gemm(xinbf, w_out + (size_t)i * DM * DI, h, h,
                    NROW, DM, DI, stream);
    }

    // --- lin2 + sigmoid ---
    lin2_kernel<<<(NROW + 7) / 8, 256, 0, stream>>>(h, lin2_w, lin2_b, out);
    (void)in_sizes; (void)n_in; (void)out_size; (void)ws_size;
}
